// GlobalGNN_74302934220861
// MI455X (gfx1250) — matrix operand order repaired and checked by reading, not compiled, not run on hardware
//
#include <hip/hip_runtime.h>

#define H   128
#define H3  384

typedef __attribute__((ext_vector_type(16))) _Float16 v16h;
typedef __attribute__((ext_vector_type(8)))  float    v8f;

union V8F { v8f v; float f[8]; };

__device__ inline void st4h(_Float16* p, float4 v) {
    union { _Float16 h[4]; unsigned long long u; } t;
    t.h[0] = (_Float16)v.x; t.h[1] = (_Float16)v.y;
    t.h[2] = (_Float16)v.z; t.h[3] = (_Float16)v.w;
    *(unsigned long long*)p = t.u;
}

// ---- CDNA5 async global->LDS copy (ASYNCcnt path) -------------------------
// Each lane copies 16 bytes: LDS[lds_off] = MEM[gaddr..gaddr+15].
__device__ inline void async_ld_b128(unsigned int lds_off, const void* gaddr) {
    asm volatile("global_load_async_to_lds_b128 %0, %1, off"
                 :: "v"(lds_off), "v"(gaddr)
                 : "memory");
}
__device__ inline void wait_async0() {
    asm volatile("s_wait_asynccnt 0x0" ::: "memory");
}

// ---------------------------------------------------------------------------
// Pack W[Out,In] (row-major, y = x @ W.T) into WMMA B tiles.
// Tile (nt, kc) holds B[k=0..31][n=0..15] = W[nt*16+n][kc*32+k] at
//   dst[(nt*(In/32)+kc)*512 + (k/16)*256 + n*16 + (k%16)]
// so lane l loads 16 contiguous f16 (one v16h) = B[k=16*(l/16)+s][n=l%16].
// ---------------------------------------------------------------------------
__global__ void pack_weights(const float* __restrict__ W, _Float16* __restrict__ dst,
                             int In, int total) {
    int idx = blockIdx.x * blockDim.x + threadIdx.x;
    if (idx >= total) return;
    int n    = (idx >> 4) & 15;
    int k    = ((idx >> 8) & 1) * 16 + (idx & 15);
    int rest = idx >> 9;
    int kcN  = In >> 5;
    int kc   = rest % kcN;
    int nt   = rest / kcN;
    dst[idx] = (_Float16)W[(size_t)(nt * 16 + n) * In + (kc * 32 + k)];
}

// i_t = ones(1,256) @ lin_w.T + lin_b  (constant row)
__global__ void compute_it(const float* __restrict__ lin_w, const float* __restrict__ lin_b,
                           float* __restrict__ i_t) {
    int o = threadIdx.x;          // 128 threads
    float s = lin_b[o];
    for (int c = 0; c < 2 * H; ++c) s += lin_w[o * 2 * H + c];
    i_t[o] = s;
}

// ---------------------------------------------------------------------------
// TiConv edge kernel: one wave (32 lanes) handles 16 edges.
//   k = [x[src], j_t, user], q = [x[dst], i_t, user]
//   h = LN( (k@wv.T)@ffn_w.T + ffn_b + q ) ; agg[dst] += h
// ---------------------------------------------------------------------------
__global__ __launch_bounds__(32)
void edge_kernel(const float* __restrict__ x, const float* __restrict__ emb,
                 const float* __restrict__ attr_time, const int* __restrict__ attr_index,
                 const int* __restrict__ src, const int* __restrict__ dst,
                 const int* __restrict__ eid, int nE,
                 const _Float16* __restrict__ wvT, const _Float16* __restrict__ ffnT,
                 const _Float16* __restrict__ linT,
                 const float* __restrict__ ffn_b, const float* __restrict__ ln_g,
                 const float* __restrict__ ln_b,  const float* __restrict__ freq,
                 const float* __restrict__ lin_b, const float* __restrict__ i_t,
                 float* __restrict__ agg) {
    __shared__ alignas(32) _Float16 kmat[16 * H3];   // 12 KB: k-matrix (f16)
    __shared__ alignas(32) _Float16 vmat[16 * H3];   // 12 KB: time tile, then v (f16)
    __shared__ alignas(16) float    qh[16 * H3];     // 24 KB: q, then h (f32)
    __shared__ int   ssrc[16], sdst[16], suid[16];
    __shared__ float st0[16], st1[16];

    const int lane = threadIdx.x;
    const int e0   = blockIdx.x * 16;

    if (lane < 16) {
        int e = e0 + lane;
        if (e < nE) {
            int id = eid[e];
            ssrc[lane] = src[e];
            sdst[lane] = dst[e];
            suid[lane] = attr_index[id];
            st0[lane]  = attr_time[2 * id + 0];
            st1[lane]  = attr_time[2 * id + 1];
        } else {
            ssrc[lane] = 0; sdst[lane] = -1; suid[lane] = 0;
            st0[lane] = 0.f; st1[lane] = 0.f;
        }
    }
    __syncthreads();

    // ---- gathers ----------------------------------------------------------
    // q[:,0:128] = x[dst] and q[:,256:384] = user stream global->LDS via the
    // CDNA5 async path (ASYNCcnt), overlapping the cos() + j_t GEMM below.
    // k[:,0:128] = x[src] needs an f32->f16 convert, so it goes via VGPRs.
    const float4 z4 = make_float4(0.f, 0.f, 0.f, 0.f);
    const float4 iv = ((const float4*)i_t)[lane];          // i_t cols 4*lane..
    for (int r = 0; r < 16; ++r) {
        bool valid = sdst[r] >= 0;
        int sr = ssrc[r], ur = suid[r], dr = valid ? sdst[r] : 0;
        async_ld_b128((unsigned int)(size_t)&qh[r * H3 + 4 * lane],
                      x + (size_t)dr * H + 4 * lane);
        async_ld_b128((unsigned int)(size_t)&qh[r * H3 + 256 + 4 * lane],
                      emb + (size_t)ur * H + 4 * lane);
        *(float4*)&qh[r * H3 + 128 + 4 * lane] = valid ? iv : z4;  // q[:,128:256]=i_t
        float4 xs = valid ? ((const float4*)(x + (size_t)sr * H))[lane] : z4;
        st4h(&kmat[r * H3 + 4 * lane], xs);                // k[:,0:128] = x[src]
    }

    // ---- time features (reuse vmat as 16x256 f16 tile) ---------------------
    _Float16* tt = vmat;
    for (int r = 0; r < 16; ++r) {
        float t0 = st0[r], t1 = st1[r];
        #pragma unroll
        for (int j = 0; j < 8; ++j) {
            int c   = lane + 32 * j;                       // 0..255
            float t = (c < H) ? t0 : t1;
            tt[r * 256 + c] = (_Float16)cosf(t * freq[c & (H - 1)]);
        }
    }
    __syncthreads();

    const int row   = lane & 15;
    const int khalf = (lane >> 4) * 16;
    const int mhalf = (lane >> 4) * 8;
    const int ncol  = lane & 15;

    // ---- j_t = timefeat @ lin_w.T + lin_b  ->  kmat[:,128:256] (f16) -------
    for (int nt = 0; nt < 8; ++nt) {
        int col = nt * 16 + ncol;
        V8F acc;
        float bias = lin_b[col];
        #pragma unroll
        for (int i = 0; i < 8; ++i) acc.f[i] = bias;
        for (int kc = 0; kc < 8; ++kc) {
            v16h a = *(const v16h*)&tt[row * 256 + kc * 32 + khalf];
            v16h b = *(const v16h*)&linT[(size_t)(nt * 8 + kc) * 512 + lane * 16];
            acc.v = __builtin_amdgcn_wmma_f32_16x16x32_f16(false, a, false, b,
                                                           (short)0, acc.v, false, false);
        }
        #pragma unroll
        for (int i = 0; i < 8; ++i)
            kmat[(i + mhalf) * H3 + 128 + col] = (_Float16)acc.f[i];
    }

    // ---- async q-copies done; build k[:,256:384] = (f16)user from LDS ------
    wait_async0();
    __syncthreads();
    for (int r = 0; r < 16; ++r) {
        float4 us = *(const float4*)&qh[r * H3 + 256 + 4 * lane];
        st4h(&kmat[r * H3 + 256 + 4 * lane], us);
    }
    __syncthreads();

    // ---- v = k @ wv.T  ->  vmat (f16) --------------------------------------
    for (int nt = 0; nt < 24; ++nt) {
        V8F acc;
        #pragma unroll
        for (int i = 0; i < 8; ++i) acc.f[i] = 0.f;
        for (int kc = 0; kc < 12; ++kc) {
            v16h a = *(const v16h*)&kmat[row * H3 + kc * 32 + khalf];
            v16h b = *(const v16h*)&wvT[(size_t)(nt * 12 + kc) * 512 + lane * 16];
            acc.v = __builtin_amdgcn_wmma_f32_16x16x32_f16(false, a, false, b,
                                                           (short)0, acc.v, false, false);
        }
        int col = nt * 16 + ncol;
        #pragma unroll
        for (int i = 0; i < 8; ++i)
            vmat[(i + mhalf) * H3 + col] = (_Float16)acc.f[i];
    }
    __syncthreads();

    // ---- h = v @ ffn_w.T + ffn_b + q  ->  qh (f32, overwrite) --------------
    for (int nt = 0; nt < 24; ++nt) {
        int col = nt * 16 + ncol;
        V8F acc;
        float fb = ffn_b[col];
        #pragma unroll
        for (int i = 0; i < 8; ++i) acc.f[i] = fb + qh[(i + mhalf) * H3 + col];
        for (int kc = 0; kc < 12; ++kc) {
            v16h a = *(const v16h*)&vmat[row * H3 + kc * 32 + khalf];
            v16h b = *(const v16h*)&ffnT[(size_t)(nt * 12 + kc) * 512 + lane * 16];
            acc.v = __builtin_amdgcn_wmma_f32_16x16x32_f16(false, a, false, b,
                                                           (short)0, acc.v, false, false);
        }
        #pragma unroll
        for (int i = 0; i < 8; ++i)
            qh[(i + mhalf) * H3 + col] = acc.f[i];
    }
    __syncthreads();

    // ---- LayerNorm per edge row + scatter-add into agg[dst] ----------------
    for (int r = 0; r < 16; ++r) {
        float s = 0.f, s2 = 0.f;
        #pragma unroll
        for (int j = 0; j < 12; ++j) {
            float v = qh[r * H3 + lane + 32 * j];
            s += v; s2 += v * v;
        }
        #pragma unroll
        for (int off = 16; off > 0; off >>= 1) {
            s  += __shfl_xor(s,  off, 32);
            s2 += __shfl_xor(s2, off, 32);
        }
        float mu   = s * (1.0f / H3);
        float var  = s2 * (1.0f / H3) - mu * mu;
        float rstd = rsqrtf(var + 1e-5f);
        int d = sdst[r];
        if (d >= 0) {
            #pragma unroll
            for (int j = 0; j < 12; ++j) {
                int c = lane + 32 * j;
                float v = (qh[r * H3 + c] - mu) * rstd * ln_g[c] + ln_b[c];
                atomicAdd(&agg[(size_t)d * H3 + c], v);
            }
        }
    }
}

// ---------------------------------------------------------------------------
// Merge MLP over all nodes: g1 = leaky( concat(x, agg) @ mw1.T + mb1 ) @ mw2.T + mb2
// agg is only stored for rows < S1 (dst range); others are exactly zero.
// ---------------------------------------------------------------------------
__global__ __launch_bounds__(32)
void merge_kernel(const float* __restrict__ x, const float* __restrict__ agg,
                  const _Float16* __restrict__ mw1T, const float* __restrict__ mb1,
                  const _Float16* __restrict__ mw2T, const float* __restrict__ mb2,
                  float* __restrict__ g1, int nNodes, int S1v) {
    __shared__ alignas(32) _Float16 amat[16 * 512];  // 16 KB: [x, agg] f16
    __shared__ alignas(32) _Float16 mmat[16 * 128];  // 4 KB: hidden f16

    const int lane = threadIdx.x;
    const int n0   = blockIdx.x * 16;
    const float4 z4 = make_float4(0.f, 0.f, 0.f, 0.f);

    for (int r = 0; r < 16; ++r) {
        int node = n0 + r;
        bool valid = node < nNodes;
        int nc = valid ? node : 0;
        float4 xs = valid ? ((const float4*)(x + (size_t)nc * H))[lane] : z4;
        st4h(&amat[r * 512 + 4 * lane], xs);
        bool ha = valid && node < S1v;
        #pragma unroll
        for (int j = 0; j < 3; ++j) {
            float4 av = ha ? ((const float4*)(agg + (size_t)nc * H3))[lane + 32 * j] : z4;
            st4h(&amat[r * 512 + 128 + 4 * (lane + 32 * j)], av);
        }
    }
    __syncthreads();

    const int row   = lane & 15;
    const int khalf = (lane >> 4) * 16;
    const int mhalf = (lane >> 4) * 8;
    const int ncol  = lane & 15;

    for (int nt = 0; nt < 8; ++nt) {
        int col = nt * 16 + ncol;
        V8F acc;
        float bias = mb1[col];
        #pragma unroll
        for (int i = 0; i < 8; ++i) acc.f[i] = bias;
        for (int kc = 0; kc < 16; ++kc) {
            v16h a = *(const v16h*)&amat[row * 512 + kc * 32 + khalf];
            v16h b = *(const v16h*)&mw1T[(size_t)(nt * 16 + kc) * 512 + lane * 16];
            acc.v = __builtin_amdgcn_wmma_f32_16x16x32_f16(false, a, false, b,
                                                           (short)0, acc.v, false, false);
        }
        #pragma unroll
        for (int i = 0; i < 8; ++i) {
            float v = acc.f[i];
            v = (v > 0.f) ? v : 0.1f * v;                  // LeakyReLU(0.1)
            mmat[(i + mhalf) * 128 + col] = (_Float16)v;
        }
    }
    __syncthreads();

    for (int nt = 0; nt < 8; ++nt) {
        int col = nt * 16 + ncol;
        V8F acc;
        float bias = mb2[col];
        #pragma unroll
        for (int i = 0; i < 8; ++i) acc.f[i] = bias;
        for (int kc = 0; kc < 4; ++kc) {
            v16h a = *(const v16h*)&mmat[row * 128 + kc * 32 + khalf];
            v16h b = *(const v16h*)&mw2T[(size_t)(nt * 4 + kc) * 512 + lane * 16];
            acc.v = __builtin_amdgcn_wmma_f32_16x16x32_f16(false, a, false, b,
                                                           (short)0, acc.v, false, false);
        }
        #pragma unroll
        for (int i = 0; i < 8; ++i) {
            int node = n0 + i + mhalf;
            if (node < nNodes) g1[(size_t)node * H + col] = acc.f[i];
        }
    }
}

// ---------------------------------------------------------------------------
// SAGE scatter: s[dst] += g1[src] ; cnt[dst] += 1
// ---------------------------------------------------------------------------
__global__ __launch_bounds__(128)
void sage_scatter(const float* __restrict__ g1, const int* __restrict__ src,
                  const int* __restrict__ dst, int nE,
                  float* __restrict__ s, float* __restrict__ cnt) {
    int e = blockIdx.x;
    if (e >= nE) return;
    int sr = src[e], d = dst[e];
    int c = threadIdx.x;
    atomicAdd(&s[(size_t)d * H + c], g1[(size_t)sr * H + c]);
    if (c == 0) atomicAdd(&cnt[d], 1.0f);
}

// ---------------------------------------------------------------------------
// SAGE out: out = normalize( mean @ wl.T + bl + g1[:S1] @ wr.T )
// ---------------------------------------------------------------------------
__global__ __launch_bounds__(32)
void sage_out(const float* __restrict__ s, const float* __restrict__ cnt,
              const float* __restrict__ g1,
              const _Float16* __restrict__ wlT, const float* __restrict__ bl,
              const _Float16* __restrict__ wrT,
              float* __restrict__ out, int nRows) {
    __shared__ alignas(32) _Float16 ma[16 * 128];   // mean (f16)
    __shared__ alignas(32) _Float16 mr[16 * 128];   // g1[:S1] (f16)
    __shared__ alignas(16) float    ob[16 * 128];

    const int lane = threadIdx.x;
    const int r0   = blockIdx.x * 16;
    const float4 z4 = make_float4(0.f, 0.f, 0.f, 0.f);

    for (int r = 0; r < 16; ++r) {
        int rowi = r0 + r;
        bool valid = rowi < nRows;
        int rc = valid ? rowi : 0;
        float inv = 1.0f / fmaxf(cnt[rc], 1.0f);
        float4 sv = valid ? ((const float4*)(s  + (size_t)rc * H))[lane] : z4;
        float4 gv = valid ? ((const float4*)(g1 + (size_t)rc * H))[lane] : z4;
        sv.x *= inv; sv.y *= inv; sv.z *= inv; sv.w *= inv;
        st4h(&ma[r * 128 + 4 * lane], sv);
        st4h(&mr[r * 128 + 4 * lane], gv);
    }
    __syncthreads();

    const int row   = lane & 15;
    const int khalf = (lane >> 4) * 16;
    const int mhalf = (lane >> 4) * 8;
    const int ncol  = lane & 15;

    for (int nt = 0; nt < 8; ++nt) {
        int col = nt * 16 + ncol;
        V8F acc;
        float bias = bl[col];
        #pragma unroll
        for (int i = 0; i < 8; ++i) acc.f[i] = bias;
        for (int kc = 0; kc < 4; ++kc) {
            v16h a = *(const v16h*)&ma[row * 128 + kc * 32 + khalf];
            v16h b = *(const v16h*)&wlT[(size_t)(nt * 4 + kc) * 512 + lane * 16];
            acc.v = __builtin_amdgcn_wmma_f32_16x16x32_f16(false, a, false, b,
                                                           (short)0, acc.v, false, false);
        }
        for (int kc = 0; kc < 4; ++kc) {
            v16h a = *(const v16h*)&mr[row * 128 + kc * 32 + khalf];
            v16h b = *(const v16h*)&wrT[(size_t)(nt * 4 + kc) * 512 + lane * 16];
            acc.v = __builtin_amdgcn_wmma_f32_16x16x32_f16(false, a, false, b,
                                                           (short)0, acc.v, false, false);
        }
        #pragma unroll
        for (int i = 0; i < 8; ++i)
            ob[(i + mhalf) * 128 + col] = acc.f[i];
    }
    __syncthreads();

    for (int r = 0; r < 16; ++r) {
        int rowi = r0 + r;
        float4 v = *(const float4*)&ob[r * 128 + 4 * lane];
        float ss = v.x * v.x + v.y * v.y + v.z * v.z + v.w * v.w;
        #pragma unroll
        for (int off = 16; off > 0; off >>= 1) ss += __shfl_xor(ss, off, 32);
        float inv = 1.0f / fmaxf(sqrtf(ss), 1e-12f);
        v.x *= inv; v.y *= inv; v.z *= inv; v.w *= inv;
        if (rowi < nRows) *(float4*)&out[(size_t)rowi * H + 4 * lane] = v;
    }
}

// ---------------------------------------------------------------------------
extern "C" void kernel_launch(void* const* d_in, const int* in_sizes, int n_in,
                              void* d_out, int out_size, void* d_ws, size_t ws_size,
                              hipStream_t stream) {
    const float* x          = (const float*)d_in[0];
    const float* emb        = (const float*)d_in[1];
    const float* attr_time  = (const float*)d_in[2];
    const int*   attr_index = (const int*)  d_in[3];
    // layer-0 edge lists (d_in[4..6]) feed only the DCE'd branch: skipped.
    const int*   src1       = (const int*)  d_in[7];
    const int*   dst1       = (const int*)  d_in[8];
    const int*   eid1       = (const int*)  d_in[9];
    const float* wv         = (const float*)d_in[10];
    const float* ffn_w      = (const float*)d_in[11];
    const float* ffn_b      = (const float*)d_in[12];
    const float* ln_g       = (const float*)d_in[13];
    const float* ln_b       = (const float*)d_in[14];
    const float* mw1        = (const float*)d_in[15];
    const float* mb1        = (const float*)d_in[16];
    const float* mw2        = (const float*)d_in[17];
    const float* mb2        = (const float*)d_in[18];
    const float* freq       = (const float*)d_in[19];
    const float* lin_w      = (const float*)d_in[20];
    const float* lin_b      = (const float*)d_in[21];
    const float* wl         = (const float*)d_in[25];   // sage1_wl
    const float* bl         = (const float*)d_in[26];   // sage1_bl
    const float* wr         = (const float*)d_in[27];   // sage1_wr

    const int nNodes = in_sizes[0] / H;      // 100000
    const int nE     = in_sizes[7];          // 50000
    const int S1v    = out_size / H;         // 5000

    // ---- workspace carve-out (all 256B aligned) ---------------------------
    char* ws = (char*)d_ws;
    size_t off = 0;
    auto carve = [&](size_t bytes) {
        void* p = ws + off;
        off = (off + bytes + 255) & ~(size_t)255;
        return p;
    };
    _Float16* wvT  = (_Float16*)carve((size_t)H3 * H3 * 2);
    _Float16* ffnT = (_Float16*)carve((size_t)H3 * H3 * 2);
    _Float16* linT = (_Float16*)carve((size_t)H * 2 * H * 2);
    _Float16* mw1T = (_Float16*)carve((size_t)H * 4 * H * 2);
    _Float16* mw2T = (_Float16*)carve((size_t)H * H * 2);
    _Float16* wlT  = (_Float16*)carve((size_t)H * H * 2);
    _Float16* wrT  = (_Float16*)carve((size_t)H * H * 2);
    float*    i_t  = (float*)   carve((size_t)H * 4);
    float*    agg  = (float*)   carve((size_t)S1v * H3 * 4);
    float*    g1   = (float*)   carve((size_t)nNodes * H * 4);
    float*    ssum = (float*)   carve((size_t)S1v * H * 4);
    float*    cntb = (float*)   carve((size_t)S1v * 4);
    (void)ws_size; (void)n_in;

    // ---- weight packing + constants ---------------------------------------
    auto pack = [&](const float* W, _Float16* D, int Out, int In) {
        int total = Out * In;
        pack_weights<<<(total + 255) / 256, 256, 0, stream>>>(W, D, In, total);
    };
    pack(wv,    wvT,  H3, H3);
    pack(ffn_w, ffnT, H3, H3);
    pack(lin_w, linT, H,  2 * H);
    pack(mw1,   mw1T, H,  4 * H);
    pack(mw2,   mw2T, H,  H);
    pack(wl,    wlT,  H,  H);
    pack(wr,    wrT,  H,  H);
    compute_it<<<1, H, 0, stream>>>(lin_w, lin_b, i_t);

    // ---- zero accumulators (graph-capturable stream memsets) --------------
    hipMemsetAsync(agg,  0, (size_t)S1v * H3 * 4, stream);
    hipMemsetAsync(ssum, 0, (size_t)S1v * H  * 4, stream);
    hipMemsetAsync(cntb, 0, (size_t)S1v * 4,      stream);

    // ---- layer 1 pipeline --------------------------------------------------
    edge_kernel<<<(nE + 15) / 16, 32, 0, stream>>>(
        x, emb, attr_time, attr_index, src1, dst1, eid1, nE,
        wvT, ffnT, linT, ffn_b, ln_g, ln_b, freq, lin_b, i_t, agg);

    merge_kernel<<<(nNodes + 15) / 16, 32, 0, stream>>>(
        x, agg, mw1T, mb1, mw2T, mb2, g1, nNodes, S1v);

    sage_scatter<<<nE, H, 0, stream>>>(g1, src1, dst1, nE, ssum, cntb);

    sage_out<<<(S1v + 15) / 16, 32, 0, stream>>>(
        ssum, cntb, g1, wlT, bl, wrT, (float*)d_out, S1v);
}
